// OptimizedDualAttentionModule_89404039233600
// MI455X (gfx1250) — matrix operand order repaired and checked by reading, not compiled, not run on hardware
//
#include <hip/hip_runtime.h>
#include <hip/hip_bf16.h>
#include <math.h>

// ---------------------------------------------------------------------------
// Shapes: x[1,3,64,64]; patch branch 16 tokens x 768; pixel branch 4096 x 3,
// attention 8 heads x d=32 over N=4096.
// ---------------------------------------------------------------------------
#define IMGN   4096          // 64*64
#define SEQN   4096
#define PDIM   768
#define NHEAD  8
#define DH     32
#define QSCALE 0.17677669529663687f   // 1/sqrt(32)

typedef __attribute__((ext_vector_type(16))) _Float16 v16h;
typedef __attribute__((ext_vector_type(8)))  float    v8f;

union V16H { v16h v; _Float16 h[16]; };
struct __align__(16) H8 { _Float16 h[8]; };

__device__ __forceinline__ float gelu_exact(float x) {
    return 0.5f * x * (1.0f + erff(x * 0.70710678118654752f));
}

// ---------------- 3x3 conv, pad 1, C=3, optional exact GELU ----------------
__global__ __launch_bounds__(256)
void k_conv3(const float* __restrict__ in, const float* __restrict__ w,
             float* __restrict__ out, int apply_gelu) {
    int idx = blockIdx.x * blockDim.x + threadIdx.x;
    if (idx >= 3 * IMGN) return;
    int c = idx >> 12, y = (idx >> 6) & 63, x = idx & 63;
    float s = 0.f;
    #pragma unroll
    for (int ci = 0; ci < 3; ci++)
        #pragma unroll
        for (int dy = -1; dy <= 1; dy++) {
            int yy = y + dy; if (yy < 0 || yy > 63) continue;
            #pragma unroll
            for (int dx = -1; dx <= 1; dx++) {
                int xx = x + dx; if (xx < 0 || xx > 63) continue;
                s += in[ci * IMGN + yy * 64 + xx] *
                     w[((c * 3 + ci) * 3 + (dy + 1)) * 3 + (dx + 1)];
            }
        }
    out[idx] = apply_gelu ? gelu_exact(s) : s;
}

// ---------------- patch embedding: pos -> patches[16][768] -----------------
__global__ __launch_bounds__(256)
void k_patch_embed(const float* __restrict__ pos, const float* __restrict__ ew,
                   const float* __restrict__ eb, float* __restrict__ patches) {
    int idx = blockIdx.x * blockDim.x + threadIdx.x;  // p*768 + o
    if (idx >= 16 * PDIM) return;
    int p = idx / PDIM, o = idx % PDIM;
    int ph = p >> 2, pw = p & 3;
    float s = eb[o];
    const float* wo = ew + (size_t)o * PDIM;
    for (int c = 0; c < 3; c++)
        for (int i = 0; i < 16; i++) {
            const float* prow = pos + c * IMGN + (ph * 16 + i) * 64 + pw * 16;
            const float* wrow = wo + c * 256 + i * 16;
            #pragma unroll
            for (int j = 0; j < 16; j++) s += prow[j] * wrow[j];
        }
    patches[idx] = s;
}

// ---------------- LayerNorm over 768, one block per row --------------------
__global__ __launch_bounds__(256)
void k_ln768(const float* __restrict__ in, const float* __restrict__ g,
             const float* __restrict__ b, float* __restrict__ out) {
    __shared__ float red[256];
    int p = blockIdx.x, t = threadIdx.x;
    const float* row = in + p * PDIM;
    float s = 0.f;
    for (int i = t; i < PDIM; i += 256) s += row[i];
    red[t] = s; __syncthreads();
    for (int o = 128; o > 0; o >>= 1) { if (t < o) red[t] += red[t + o]; __syncthreads(); }
    float mu = red[0] * (1.0f / PDIM);
    __syncthreads();
    float v = 0.f;
    for (int i = t; i < PDIM; i += 256) { float d = row[i] - mu; v += d * d; }
    red[t] = v; __syncthreads();
    for (int o = 128; o > 0; o >>= 1) { if (t < o) red[t] += red[t + o]; __syncthreads(); }
    float rinv = rsqrtf(red[0] * (1.0f / PDIM) + 1e-5f);
    for (int i = t; i < PDIM; i += 256)
        out[p * PDIM + i] = (row[i] - mu) * rinv * g[i] + b[i];
}

// ---------------- generic small matmul: C[M,N] = A[M,K] @ B[K,N] (+bias) ---
__global__ __launch_bounds__(256)
void k_matmul(const float* __restrict__ A, const float* __restrict__ B,
              const float* __restrict__ bias, float* __restrict__ C,
              int M, int K, int N) {
    int idx = blockIdx.x * blockDim.x + threadIdx.x;
    if (idx >= M * N) return;
    int m = idx / N, n = idx % N;
    float s = bias ? bias[n] : 0.f;
    for (int k = 0; k < K; k++) s += A[m * K + k] * B[k * N + n];
    C[idx] = s;
}

// ---------------- tiny 16-token patch attention (block per head) -----------
__global__ __launch_bounds__(256)
void k_patch_attn(const float* __restrict__ qkvp, float* __restrict__ op) {
    int h = blockIdx.x, t = threadIdx.x;
    __shared__ float sc[16][16];
    int q = t >> 4, k = t & 15;
    float s = 0.f;
    #pragma unroll
    for (int d = 0; d < DH; d++)
        s += qkvp[q * PDIM + h * DH + d] * qkvp[k * PDIM + 256 + h * DH + d];
    sc[q][k] = s * QSCALE;
    __syncthreads();
    if (k == 0) {
        float mx = -1e30f;
        for (int j = 0; j < 16; j++) mx = fmaxf(mx, sc[q][j]);
        float sum = 0.f;
        for (int j = 0; j < 16; j++) { float e = __expf(sc[q][j] - mx); sc[q][j] = e; sum += e; }
        float inv = 1.f / sum;
        for (int j = 0; j < 16; j++) sc[q][j] *= inv;
    }
    __syncthreads();
    for (int o = t; o < 512; o += 256) {
        int qq = o >> 5, d = o & 31;
        float acc = 0.f;
        for (int j = 0; j < 16; j++)
            acc += sc[qq][j] * qkvp[j * PDIM + 512 + h * DH + d];
        op[qq * 256 + h * DH + d] = acc;
    }
}

// ---------------- patch reconstruction (conv-transpose k=stride=16) --------
__global__ __launch_bounds__(256)
void k_recon(const float* __restrict__ pp, const float* __restrict__ rw,
             const float* __restrict__ rb, float* __restrict__ combined) {
    int idx = blockIdx.x * blockDim.x + threadIdx.x;  // d*4096 + y*64 + x
    if (idx >= 3 * IMGN) return;
    int d = idx >> 12, y = (idx >> 6) & 63, x = idx & 63;
    int p = (y >> 4) * 4 + (x >> 4), i = y & 15, j = x & 15;
    float s = rb[d];
    const float* pr = pp + p * PDIM;
    for (int c = 0; c < PDIM; c++)
        s += pr[c] * rw[c * PDIM + d * 256 + i * 16 + j];
    combined[idx] = s;
}

// ---------------- pixel branch: fused LN + QKV, f16 in WMMA layouts --------
// qh: [h][n][32] row-major (pre-scaled by 1/sqrt(32)); kh: [h][n][32]; vt: [h][32][n]
__global__ __launch_bounds__(256)
void k_seq_qkv(const float* __restrict__ pos, const float* __restrict__ g2,
               const float* __restrict__ b2, const float* __restrict__ W,
               _Float16* __restrict__ qh, _Float16* __restrict__ kh,
               _Float16* __restrict__ vt) {
    int idx = blockIdx.x * blockDim.x + threadIdx.x;  // n*768 + col
    if (idx >= SEQN * PDIM) return;
    int n = idx / PDIM, col = idx % PDIM;
    float c0 = pos[n], c1 = pos[IMGN + n], c2 = pos[2 * IMGN + n];
    float mu = (c0 + c1 + c2) * (1.f / 3.f);
    float d0 = c0 - mu, d1 = c1 - mu, d2 = c2 - mu;
    float rinv = rsqrtf((d0 * d0 + d1 * d1 + d2 * d2) * (1.f / 3.f) + 1e-5f);
    float s0 = d0 * rinv * g2[0] + b2[0];
    float s1 = d1 * rinv * g2[1] + b2[1];
    float s2 = d2 * rinv * g2[2] + b2[2];
    float val = s0 * W[col] + s1 * W[PDIM + col] + s2 * W[2 * PDIM + col];
    if (col < 256) {
        int h = col >> 5, d = col & 31;
        qh[((size_t)h * SEQN + n) * DH + d] = (_Float16)(val * QSCALE);
    } else if (col < 512) {
        int cc = col - 256; int h = cc >> 5, d = cc & 31;
        kh[((size_t)h * SEQN + n) * DH + d] = (_Float16)val;
    } else {
        int cc = col - 512; int h = cc >> 5, d = cc & 31;
        vt[((size_t)h * DH + d) * SEQN + n] = (_Float16)val;
    }
}

// ---------------- flash attention: 1 wave per (head, 32-query tile) --------
// Per iteration: 64 keys.  8 S-WMMAs (v_wmma_f32_16x16x32_f16, K-dim == head
// dim 32), online softmax with one butterfly per row per 64 keys, P re-enters
// A-operand layout via 4KB LDS, 8 P.V WMMAs accumulating f32 fragments.
// K/V B-fragments are each reused by two Q row-tiles.
__global__ __launch_bounds__(32)
void k_flash_attn(const _Float16* __restrict__ qh, const _Float16* __restrict__ kh,
                  const _Float16* __restrict__ vt, float* __restrict__ O) {
    const int head = blockIdx.y;
    const int qb   = blockIdx.x;          // 32 queries per block
    const int lane = threadIdx.x;
    const int hf   = lane >> 4;           // half-wave: 0 or 1
    const int l15  = lane & 15;

    const _Float16* qbase = qh + (((size_t)head * SEQN + qb * 32) * DH);
    const _Float16* kbase = kh + ((size_t)head * SEQN * DH);
    const _Float16* vbase = vt + ((size_t)head * DH * SEQN);

    // A operands (16x32; this code's contiguous k order, same as B): lane row m=l15; halves 0..15 hold
    // K=hf*16..+15 => two 16B chunks.
    V16H aq[2];
    #pragma unroll
    for (int t = 0; t < 2; t++) {
        const _Float16* r = qbase + (t * 16 + l15) * DH + hf * 16;
        *(H8*)&aq[t].h[0] = *(const H8*)(r);
        *(H8*)&aq[t].h[8] = *(const H8*)(r + 8);
    }

    v8f acc[2][2] = {};                    // [row-tile][d-half]
    float m_run[2][8], l_run[2][8];
    #pragma unroll
    for (int t = 0; t < 2; t++)
        #pragma unroll
        for (int r = 0; r < 8; r++) { m_run[t][r] = -1e30f; l_run[t][r] = 0.f; }

    __shared__ _Float16 plds[32 * 64];     // P tile: 32 queries x 64 keys

    for (int kb = 0; kb < SEQN; kb += 64) {
        // ---- 4 K B-fragments: keys kb+c*16+l15, d = hf*16+e (32B/lane) ----
        V16H bk[4];
        #pragma unroll
        for (int c = 0; c < 4; c++) {
            const _Float16* r = kbase + (size_t)(kb + c * 16 + l15) * DH + hf * 16;
            *(H8*)&bk[c].h[0] = *(const H8*)(r);
            *(H8*)&bk[c].h[8] = *(const H8*)(r + 8);
        }
        int kbn = (kb + 64 < SEQN) ? kb + 64 : kb;   // prefetch next block
        __builtin_prefetch(kbase + (size_t)(kbn + l15) * DH, 0, 1);
        __builtin_prefetch(vbase + (size_t)l15 * SEQN + kbn, 0, 1);

        // ---- S tiles + online softmax; exp straight into LDS as f16 -------
        #pragma unroll
        for (int t = 0; t < 2; t++) {
            v8f zero = {};
            v8f s0 = __builtin_amdgcn_wmma_f32_16x16x32_f16(false, aq[t].v, false, bk[0].v,
                                                            (short)0, zero, false, false);
            v8f s1 = __builtin_amdgcn_wmma_f32_16x16x32_f16(false, aq[t].v, false, bk[1].v,
                                                            (short)0, zero, false, false);
            v8f s2 = __builtin_amdgcn_wmma_f32_16x16x32_f16(false, aq[t].v, false, bk[2].v,
                                                            (short)0, zero, false, false);
            v8f s3 = __builtin_amdgcn_wmma_f32_16x16x32_f16(false, aq[t].v, false, bk[3].v,
                                                            (short)0, zero, false, false);
            #pragma unroll
            for (int r = 0; r < 8; r++) {
                // row m = r + hf*8 of tile t; its 64 scores live across the
                // 16 lanes of this half (4 fragments x 16 lanes)
                float mx = fmaxf(fmaxf(s0[r], s1[r]), fmaxf(s2[r], s3[r]));
                #pragma unroll
                for (int sh = 1; sh < 16; sh <<= 1) mx = fmaxf(mx, __shfl_xor(mx, sh, 32));
                float mnew = fmaxf(m_run[t][r], mx);
                float corr = __expf(m_run[t][r] - mnew);
                m_run[t][r] = mnew;
                float e0 = __expf(s0[r] - mnew);
                float e1 = __expf(s1[r] - mnew);
                float e2 = __expf(s2[r] - mnew);
                float e3 = __expf(s3[r] - mnew);
                int row = t * 16 + r + hf * 8;
                plds[row * 64 +      l15] = (_Float16)e0;
                plds[row * 64 + 16 + l15] = (_Float16)e1;
                plds[row * 64 + 32 + l15] = (_Float16)e2;
                plds[row * 64 + 48 + l15] = (_Float16)e3;
                float ps = (e0 + e1) + (e2 + e3);
                #pragma unroll
                for (int sh = 1; sh < 16; sh <<= 1) ps += __shfl_xor(ps, sh, 32);
                l_run[t][r] = l_run[t][r] * corr + ps;
                acc[t][0][r] *= corr;
                acc[t][1][r] *= corr;
            }
        }
        __syncthreads();

        // ---- 4 V B-fragments: d = j*16+l15, keys kb+ch*32+hf*16+e ---------
        V16H bv[2][2];                     // [key-chunk][d-half]
        #pragma unroll
        for (int ch = 0; ch < 2; ch++)
            #pragma unroll
            for (int j = 0; j < 2; j++) {
                const _Float16* cp = vbase + (size_t)(j * 16 + l15) * SEQN
                                   + kb + ch * 32 + hf * 16;
                *(H8*)&bv[ch][j].h[0] = *(const H8*)(cp);
                *(H8*)&bv[ch][j].h[8] = *(const H8*)(cp + 8);
            }

        // ---- P.V: reload P in A layout, 8 accumulating WMMAs --------------
        #pragma unroll
        for (int t = 0; t < 2; t++) {
            #pragma unroll
            for (int ch = 0; ch < 2; ch++) {
                V16H ap;
                const _Float16* pr = &plds[(t * 16 + l15) * 64 + ch * 32 + hf * 16];
                *(H8*)&ap.h[0] = *(const H8*)(pr);
                *(H8*)&ap.h[8] = *(const H8*)(pr + 8);
                acc[t][0] = __builtin_amdgcn_wmma_f32_16x16x32_f16(false, ap.v, false, bv[ch][0].v,
                                                                   (short)0, acc[t][0], false, false);
                acc[t][1] = __builtin_amdgcn_wmma_f32_16x16x32_f16(false, ap.v, false, bv[ch][1].v,
                                                                   (short)0, acc[t][1], false, false);
            }
        }
        __syncthreads();  // before next iteration overwrites plds
    }

    // epilogue: O[h][qb*32 + t*16 + m][d] = acc / l
    float* obase = O + ((size_t)head * SEQN + qb * 32) * DH;
    #pragma unroll
    for (int t = 0; t < 2; t++)
        #pragma unroll
        for (int r = 0; r < 8; r++) {
            float inv = 1.0f / l_run[t][r];
            int row = t * 16 + r + hf * 8;
            obase[row * DH + l15]      = acc[t][0][r] * inv;
            obase[row * DH + 16 + l15] = acc[t][1][r] * inv;
        }
}

// ---------------- pixel-branch output projection (256 -> 3), add in --------
__global__ __launch_bounds__(256)
void k_seq_proj(const float* __restrict__ O, const float* __restrict__ W,
                const float* __restrict__ b, float* __restrict__ combined) {
    int idx = blockIdx.x * blockDim.x + threadIdx.x;  // n*3 + c
    if (idx >= SEQN * 3) return;
    int n = idx / 3, c = idx % 3;
    float s = b[c];
    for (int h = 0; h < NHEAD; h++) {
        const float* orow = O + ((size_t)h * SEQN + n) * DH;
        #pragma unroll
        for (int d = 0; d < DH; d++)
            s += orow[d] * W[(h * DH + d) * 3 + c];
    }
    combined[c * IMGN + n] += s;   // combined already holds out_patch
}

// ---------------------------------------------------------------------------
extern "C" void kernel_launch(void* const* d_in, const int* in_sizes, int n_in,
                              void* d_out, int out_size, void* d_ws, size_t ws_size,
                              hipStream_t stream) {
    const float* x        = (const float*)d_in[0];
    const float* pe_w1    = (const float*)d_in[1];
    const float* pe_w2    = (const float*)d_in[2];
    const float* embed_w  = (const float*)d_in[3];
    const float* embed_b  = (const float*)d_in[4];
    const float* ln1_g    = (const float*)d_in[5];
    const float* ln1_b    = (const float*)d_in[6];
    const float* qkv_p_w  = (const float*)d_in[7];
    const float* proj_p_w = (const float*)d_in[8];
    const float* proj_p_b = (const float*)d_in[9];
    const float* recon_w  = (const float*)d_in[10];
    const float* recon_b  = (const float*)d_in[11];
    const float* ln2_g    = (const float*)d_in[12];
    const float* ln2_b    = (const float*)d_in[13];
    const float* qkv_s_w  = (const float*)d_in[14];
    const float* proj_s_w = (const float*)d_in[15];
    const float* proj_s_b = (const float*)d_in[16];
    const float* pd_w1    = (const float*)d_in[17];
    const float* pd_w2    = (const float*)d_in[18];
    float* out = (float*)d_out;

    // ---- workspace carve-up ----
    float* ws = (float*)d_ws;
    float* t1       = ws;                 ws += 3 * IMGN;       // 12288
    float* pos      = ws;                 ws += 3 * IMGN;
    float* patches  = ws;                 ws += 16 * PDIM;
    float* pn       = ws;                 ws += 16 * PDIM;
    float* qkvp     = ws;                 ws += 16 * PDIM;
    float* op       = ws;                 ws += 16 * 256;
    float* pp       = ws;                 ws += 16 * PDIM;
    float* combined = ws;                 ws += 3 * IMGN;
    float* t2       = ws;                 ws += 3 * IMGN;
    float* Obuf     = ws;                 ws += (size_t)NHEAD * SEQN * DH;  // 4MB
    _Float16* qh = (_Float16*)ws;
    _Float16* kh = qh + (size_t)NHEAD * SEQN * DH;
    _Float16* vt = kh + (size_t)NHEAD * SEQN * DH;

    const int B256 = 256;
    const int gPix = (3 * IMGN + B256 - 1) / B256;          // 48

    // positional encoder: conv -> gelu -> conv
    k_conv3<<<gPix, B256, 0, stream>>>(x, pe_w1, t1, 1);
    k_conv3<<<gPix, B256, 0, stream>>>(t1, pe_w2, pos, 0);

    // ---- patch branch ----
    k_patch_embed<<<(16 * PDIM + 255) / 256, B256, 0, stream>>>(pos, embed_w, embed_b, patches);
    k_ln768<<<16, B256, 0, stream>>>(patches, ln1_g, ln1_b, pn);
    k_matmul<<<(16 * PDIM + 255) / 256, B256, 0, stream>>>(pn, qkv_p_w, nullptr, qkvp, 16, PDIM, PDIM);
    k_patch_attn<<<NHEAD, B256, 0, stream>>>(qkvp, op);
    k_matmul<<<(16 * PDIM + 255) / 256, B256, 0, stream>>>(op, proj_p_w, proj_p_b, pp, 16, 256, PDIM);
    k_recon<<<gPix, B256, 0, stream>>>(pp, recon_w, recon_b, combined);

    // ---- pixel branch: LN+QKV into WMMA-friendly f16 layouts ----
    k_seq_qkv<<<(SEQN * PDIM + 255) / 256, B256, 0, stream>>>(pos, ln2_g, ln2_b, qkv_s_w, qh, kh, vt);
    dim3 fg(SEQN / 32, NHEAD);
    k_flash_attn<<<fg, 32, 0, stream>>>(qh, kh, vt, Obuf);
    k_seq_proj<<<(SEQN * 3 + 255) / 256, B256, 0, stream>>>(Obuf, proj_s_w, proj_s_b, combined);

    // patch decoder: conv -> gelu -> conv
    k_conv3<<<gPix, B256, 0, stream>>>(combined, pd_w1, t2, 1);
    k_conv3<<<gPix, B256, 0, stream>>>(t2, pd_w2, out, 0);
}
